// Ml4fTransformer_48421461295652
// MI455X (gfx1250) — matrix operand order repaired and checked by reading, not compiled, not run on hardware
//
#include <hip/hip_runtime.h>
#include <hip/hip_bf16.h>
#include <cstdint>

// ---------------------------------------------------------------------------
// Problem dims (from reference)
// ---------------------------------------------------------------------------
#define BATCH 16
#define CTXS  512
#define PREDS 64
#define DE    512
#define NHEAD 8
#define NLAY  4
#define HDE   (NHEAD * DE)      // 4096
#define FFD   (10 * DE)         // 5120
#define MROWS (BATCH * CTXS)    // 8192

typedef __attribute__((ext_vector_type(16))) __bf16 bf16x16;
typedef __attribute__((ext_vector_type(8)))  float  f32x8;

// Use gfx1250 async global->LDS DMA (ASYNCcnt) for tile staging.
#define USE_ASYNC 1

#if USE_ASYNC
#define ASYNC_WAIT(imm) asm volatile("s_wait_asynccnt " #imm ::: "memory")
#else
#define ASYNC_WAIT(imm)
#endif

// One 16-byte global -> LDS transfer per call.
// Async path: global_load_async_to_lds_b128 (VDST = per-lane LDS byte address,
// VADDR = 64-bit global address), tracked by ASYNCcnt, no VGPR round trip.
__device__ __forceinline__ void copy_b128(const __bf16* gp, __bf16* lp)
{
#if USE_ASYNC
    unsigned lds_off = (unsigned)(uintptr_t)lp;           // addr[31:0] = LDS offset
    unsigned long long ga = (unsigned long long)(uintptr_t)gp;
    asm volatile("global_load_async_to_lds_b128 %0, %1, off"
                 :: "v"(lds_off), "v"(ga) : "memory");
#else
    *(uint4*)lp = *(const uint4*)gp;
#endif
}

// ---------------------------------------------------------------------------
// Generic batched WMMA GEMM, bf16 inputs, fp32 accumulate.
//   C = A @ B (+bias)(+resid)(relu?), optional bf16 and/or fp32 output.
//   Block tile 128x128, K-step 32, 8 waves (wave32), each wave: 32x64 C tile.
//   Double-buffered LDS; A (and B when transB) staged with async-to-LDS DMA.
//   Out-of-range rows/cols are address-CLAMPED (not zeroed): garbage only
//   reaches C elements that are never stored, and clamping keeps every
//   access in-bounds without exec-mask branches in the hot loop.
//   Batch offsets decompose as (bb, hh) = (batch/nH, batch%nH) so per-head
//   strided views of [B,S,H*D] tensors need no transpose copies.
//   splitK>1: atomicAdd partials into Cf (bias/relu/resid/Cb must be off).
// ---------------------------------------------------------------------------
#define BM 128
#define BN 128
#define BK 32
#define LDSS 40   // 32 + 8 pad: 20-dword row stride -> conflict-free b128 reads

union FragU { uint4 u4[2]; bf16x16 v; };

__global__ void __launch_bounds__(256)
wmma_gemm_bf16(const __bf16* __restrict__ A, const __bf16* __restrict__ B,
               const float* __restrict__ bias, const float* __restrict__ resid,
               float* __restrict__ Cf, __bf16* __restrict__ Cb,
               int M, int N, int K, int lda, int ldb, int ldc,
               int nH,
               long long sAb, long long sAh,
               long long sBb, long long sBh,
               long long sCb, long long sCh,
               int relu, int transB, int splitK)
{
    __shared__ __bf16 As[2][BM * LDSS];
    __shared__ __bf16 Bs[2][BN * LDSS];

    const int tid   = threadIdx.x;
    const int ks    = (int)(blockIdx.z) % splitK;
    const int batch = (int)(blockIdx.z) / splitK;
    const int bb    = batch / nH;
    const int hh    = batch % nH;

    const __bf16* Ab = A + (long long)bb * sAb + (long long)hh * sAh;
    const __bf16* Bb = B + (long long)bb * sBb + (long long)hh * sBh;
    const long long coff = (long long)bb * sCb + (long long)hh * sCh;

    const int m0 = blockIdx.y * BM;
    const int n0 = blockIdx.x * BN;
    const int Kchunk = K / splitK;         // all uses: multiples of BK
    const int k0 = ks * Kchunk;
    const int ntiles = Kchunk / BK;

    const int wave  = tid >> 5;
    const int lane  = tid & 31;
    const int waveM = (wave & 3) * 32;     // 4 wave-rows
    const int waveN = (wave >> 2) * 64;    // 2 wave-cols
    const int lm    = lane & 15;
    const int kaA   = (lane >> 4) * 8;     // A frag: K blocks of 8 per lane-half
    const int kaB   = (lane >> 4) * 8;     // B frag: K blocks of 8 per lane-half

    f32x8 acc[2][4];
#pragma unroll
    for (int mi = 0; mi < 2; ++mi)
#pragma unroll
        for (int ni = 0; ni < 4; ++ni)
#pragma unroll
            for (int r = 0; r < 8; ++r) acc[mi][ni][r] = 0.0f;

    // ---- stage tile (kt) into LDS buffer `buf` ----
    auto stage = [&](int kt, int buf) {
        __bf16* Asb = &As[buf][0];
        __bf16* Bsb = &Bs[buf][0];
#pragma unroll
        for (int it = 0; it < 2; ++it) {               // A: BM x BK
            int idx = (it * 256 + tid) * 8;
            int r = idx >> 5, c = idx & 31;
            int gm = m0 + r; if (gm >= M) gm = M - 1;  // clamp (value is dead)
            copy_b128(Ab + (long long)gm * lda + (k0 + kt + c),
                      Asb + r * LDSS + c);
        }
        if (transB) {   // B row-major [N x K] (e.g. K-matrix of Q.K^T)
#pragma unroll
            for (int it = 0; it < 2; ++it) {
                int idx = (it * 256 + tid) * 8;
                int r = idx >> 5, c = idx & 31;
                int gn = n0 + r; if (gn >= N) gn = N - 1;
                copy_b128(Bb + (long long)gn * ldb + (k0 + kt + c),
                          Bsb + r * LDSS + c);
            }
        } else {        // B row-major [K x N]: transpose while staging (manual)
#pragma unroll
            for (int it = 0; it < 2; ++it) {
                int idx = (it * 256 + tid) * 8;
                int kk = idx >> 7, n = idx & 127;
                int gn = n0 + n; if (gn >= N) gn = N - 8;   // N % 8 == 0
                uint4 val = *(const uint4*)(Bb + (long long)(k0 + kt + kk) * ldb + gn);
                __bf16 tmp[8];
                *(uint4*)tmp = val;
#pragma unroll
                for (int j = 0; j < 8; ++j) Bsb[(n + j) * LDSS + kk] = tmp[j];
            }
        }
    };

    // ---- 8x WMMA on LDS buffer `buf` ----
    auto compute = [&](int buf) {
        const __bf16* Asb = &As[buf][0];
        const __bf16* Bsb = &Bs[buf][0];
        FragU af[2], bf_[4];
#pragma unroll
        for (int mi = 0; mi < 2; ++mi) {
            const __bf16* p = &Asb[(waveM + mi * 16 + lm) * LDSS + kaA];
            af[mi].u4[0] = *(const uint4*)p;            // K = ka .. ka+7
            af[mi].u4[1] = *(const uint4*)(p + 16);     // K = 16+ka .. 16+ka+7
        }
#pragma unroll
        for (int ni = 0; ni < 4; ++ni) {
            const __bf16* p = &Bsb[(waveN + ni * 16 + lm) * LDSS + kaB];
            bf_[ni].u4[0] = *(const uint4*)p;           // K = kb .. kb+7
            bf_[ni].u4[1] = *(const uint4*)(p + 16);    // K = 16+kb .. 16+kb+7
        }
#pragma unroll
        for (int mi = 0; mi < 2; ++mi)
#pragma unroll
            for (int ni = 0; ni < 4; ++ni)
                acc[mi][ni] = __builtin_amdgcn_wmma_f32_16x16x32_bf16(
                    false, af[mi].v, false, bf_[ni].v,
                    (short)0, acc[mi][ni], false, false);
    };

    // ---- double-buffered pipeline ----
    stage(0, 0);
    for (int t = 0; t < ntiles; ++t) {
        int cur = t & 1;
        bool more = (t + 1 < ntiles);
        if (more) stage((t + 1) * BK, cur ^ 1);
        // In-order ASYNCcnt: <=4 pending means the current tile's 4 transfers
        // per thread are complete while the next tile's stay in flight.
        if (more) { if (transB) ASYNC_WAIT(4); else ASYNC_WAIT(2); }
        else      { ASYNC_WAIT(0); }
        __syncthreads();
        compute(cur);
        __syncthreads();
    }

    // ---- epilogue (C frag: VGPR r -> M = r + 8*(lane>=16), N = lane&15) ----
    const int mlane = (lane >> 4) << 3;
    const int nlane = lane & 15;
#pragma unroll
    for (int mi = 0; mi < 2; ++mi) {
#pragma unroll
        for (int ni = 0; ni < 4; ++ni) {
            int ccol = n0 + waveN + ni * 16 + nlane;
            if (ccol >= N) continue;
#pragma unroll
            for (int r = 0; r < 8; ++r) {
                int crow = m0 + waveM + mi * 16 + mlane + r;
                if (crow >= M) continue;
                float v = acc[mi][ni][r];
                long long off = coff + (long long)crow * ldc + ccol;
                if (splitK > 1) {
                    atomicAdd(&Cf[off], v);
                } else {
                    if (bias)  v += bias[ccol];
                    if (resid) v += resid[off];
                    if (relu)  v = fmaxf(v, 0.0f);
                    if (Cf) Cf[off] = v;
                    if (Cb) Cb[off] = (__bf16)v;
                }
            }
        }
    }
}

// ---------------------------------------------------------------------------
// Support kernels
// ---------------------------------------------------------------------------
__global__ void cvt_f32_bf16_kernel(const float* __restrict__ in,
                                    __bf16* __restrict__ out, long long n)
{
    long long i = (long long)blockIdx.x * blockDim.x + threadIdx.x;
    long long stride = (long long)gridDim.x * blockDim.x;
    for (; i < n; i += stride) out[i] = (__bf16)in[i];
}

// h = x + PE(pos, i) faithful to the reference loop
__global__ void pe_add_kernel(const float* __restrict__ x, float* __restrict__ out,
                              int S, int D, long long total)
{
    long long i = (long long)blockIdx.x * blockDim.x + threadIdx.x;
    long long stride = (long long)gridDim.x * blockDim.x;
    for (; i < total; i += stride) {
        int d = (int)(i % D);
        int s = (int)((i / D) % S);
        bool even = (d & 1) == 0;
        float fi = (float)d;
        float expo = even ? (2.0f * fi / (float)D) : (2.0f * (fi + 1.0f) / (float)D);
        float ang = (float)s / powf(10000.0f, expo);
        out[i] = x[i] + (even ? sinf(ang) : cosf(ang));
    }
}

// LayerNorm over D, fp32 in -> bf16 out (feeds WMMA GEMMs). One block per row.
__global__ void __launch_bounds__(256)
ln_bf16_kernel(const float* __restrict__ x, const float* __restrict__ g,
               const float* __restrict__ b, __bf16* __restrict__ out, int D)
{
    __shared__ float red[256];
    long long row = blockIdx.x;
    const float* xr = x + row * D;
    float s = 0.0f;
    for (int i = threadIdx.x; i < D; i += 256) s += xr[i];
    red[threadIdx.x] = s; __syncthreads();
    for (int st = 128; st > 0; st >>= 1) {
        if ((int)threadIdx.x < st) red[threadIdx.x] += red[threadIdx.x + st];
        __syncthreads();
    }
    float mu = red[0] / (float)D; __syncthreads();
    float v = 0.0f;
    for (int i = threadIdx.x; i < D; i += 256) { float d = xr[i] - mu; v += d * d; }
    red[threadIdx.x] = v; __syncthreads();
    for (int st = 128; st > 0; st >>= 1) {
        if ((int)threadIdx.x < st) red[threadIdx.x] += red[threadIdx.x + st];
        __syncthreads();
    }
    float rs = rsqrtf(red[0] / (float)D + 1e-5f);
    for (int i = threadIdx.x; i < D; i += 256)
        out[row * D + i] = (__bf16)((xr[i] - mu) * rs * g[i] + b[i]);
}

// Row softmax, fp32 scores -> bf16 probs. One block per row.
__global__ void __launch_bounds__(256)
softmax_bf16_kernel(const float* __restrict__ s, __bf16* __restrict__ p, int C)
{
    __shared__ float red[256];
    long long row = blockIdx.x;
    const float* sr = s + row * C;
    float m = -3.4e38f;
    for (int i = threadIdx.x; i < C; i += 256) m = fmaxf(m, sr[i]);
    red[threadIdx.x] = m; __syncthreads();
    for (int st = 128; st > 0; st >>= 1) {
        if ((int)threadIdx.x < st)
            red[threadIdx.x] = fmaxf(red[threadIdx.x], red[threadIdx.x + st]);
        __syncthreads();
    }
    m = red[0]; __syncthreads();
    float sum = 0.0f;
    for (int i = threadIdx.x; i < C; i += 256) sum += __expf(sr[i] - m);
    red[threadIdx.x] = sum; __syncthreads();
    for (int st = 128; st > 0; st >>= 1) {
        if ((int)threadIdx.x < st) red[threadIdx.x] += red[threadIdx.x + st];
        __syncthreads();
    }
    float inv = 1.0f / red[0];
    for (int i = threadIdx.x; i < C; i += 256)
        p[row * C + i] = (__bf16)(__expf(sr[i] - m) * inv);
}

// learn epilogue: relu(partial + bias)
__global__ void bias_relu_kernel(const float* __restrict__ acc,
                                 const float* __restrict__ bias,
                                 float* __restrict__ out, int n, int nb)
{
    int i = blockIdx.x * blockDim.x + threadIdx.x;
    if (i < n) out[i] = fmaxf(acc[i] + bias[i % nb], 0.0f);
}

// ---------------- decoder (d_model == 1): tiny scalar kernels ----------------
__global__ void ln_small_kernel(const float* __restrict__ x, const float* __restrict__ g,
                                const float* __restrict__ b, float* __restrict__ out,
                                int rows, int D)
{
    int r = blockIdx.x * blockDim.x + threadIdx.x;
    if (r >= rows) return;
    const float* xr = x + (long long)r * D;
    float mu = 0.0f;
    for (int i = 0; i < D; ++i) mu += xr[i];
    mu /= (float)D;
    float var = 0.0f;
    for (int i = 0; i < D; ++i) { float d = xr[i] - mu; var += d * d; }
    var /= (float)D;
    float rs = rsqrtf(var + 1e-5f);
    for (int i = 0; i < D; ++i) out[(long long)r * D + i] = (xr[i] - mu) * rs * g[i] + b[i];
}

// Fused decoder attention (H=8 heads of dim 1, S=64). One thread per (b, q).
__global__ void dec_attn_kernel(const float* __restrict__ qsrc,
                                const float* __restrict__ kvsrc,
                                const float* __restrict__ wq, const float* __restrict__ wk,
                                const float* __restrict__ wv, const float* __restrict__ wu,
                                const float* __restrict__ bu,
                                float* __restrict__ dio, int causal)
{
    int t = blockIdx.x * blockDim.x + threadIdx.x;
    if (t >= BATCH * PREDS) return;
    int bi = t >> 6, qi = t & 63;
    float q = qsrc[t];
    const float* kvb = kvsrc + bi * 64;
    float outv = 0.0f;
    for (int h = 0; h < 8; ++h) {
        float qh = q * wq[h], wkh = wk[h], wvh = wv[h];
        float m = -3.4e38f;
        for (int j = 0; j < 64; ++j) {
            float sc = (causal && j > qi) ? -1e9f : qh * (kvb[j] * wkh);
            m = fmaxf(m, sc);
        }
        float s = 0.0f, o = 0.0f;
        for (int j = 0; j < 64; ++j) {
            float sc = (causal && j > qi) ? -1e9f : qh * (kvb[j] * wkh);
            float e = __expf(sc - m);
            s += e;
            o += e * (kvb[j] * wvh);
        }
        outv += (o / s) * wu[h];
    }
    dio[t] += outv + bu[0];
}

__global__ void dec_ffn_kernel(const float* __restrict__ d2,
                               const float* __restrict__ w1, const float* __restrict__ b1,
                               const float* __restrict__ w2, const float* __restrict__ b2,
                               float* __restrict__ dio)
{
    int t = blockIdx.x * blockDim.x + threadIdx.x;
    if (t >= BATCH * PREDS) return;
    float x = d2[t], acc = 0.0f;
    for (int u = 0; u < 10; ++u) {
        float f = fmaxf(x * w1[u] + b1[u], 0.0f);
        acc += f * w2[u];
    }
    dio[t] += acc + b2[0];
}

__global__ void map_head_kernel(const float* __restrict__ dec_out,
                                const float* __restrict__ w,
                                const float* __restrict__ bias,
                                float* __restrict__ out)
{
    int t = blockIdx.x * blockDim.x + threadIdx.x;
    if (t >= BATCH * PREDS) return;
    int bi = t >> 6, n = t & 63;
    float acc = bias[n];
    for (int k = 0; k < 64; ++k) acc += dec_out[bi * 64 + k] * w[k * 64 + n];
    out[t] = fmaxf(acc, 0.0f);
}

// ---------------------------------------------------------------------------
// Host side
// ---------------------------------------------------------------------------
static inline void launch_gemm(hipStream_t st,
    const __bf16* A, const __bf16* B, const float* bias, const float* resid,
    float* Cf, __bf16* Cb, int M, int N, int K, int lda, int ldb, int ldc,
    int batch, int nH, long long sAb, long long sAh, long long sBb, long long sBh,
    long long sCb, long long sCh, int relu, int transB, int splitK)
{
    dim3 grid((N + BN - 1) / BN, (M + BM - 1) / BM, batch * splitK);
    wmma_gemm_bf16<<<grid, 256, 0, st>>>(A, B, bias, resid, Cf, Cb, M, N, K,
                                         lda, ldb, ldc, nH, sAb, sAh, sBb, sBh,
                                         sCb, sCh, relu, transB, splitK);
}

static inline void launch_cvt(hipStream_t st, const float* in, __bf16* out, long long n)
{
    cvt_f32_bf16_kernel<<<2048, 256, 0, st>>>(in, out, n);
}

extern "C" void kernel_launch(void* const* d_in, const int* in_sizes, int n_in,
                              void* d_out, int out_size, void* d_ws, size_t ws_size,
                              hipStream_t stream)
{
    (void)in_sizes; (void)n_in; (void)out_size; (void)ws_size;
    // ---- inputs (setup_inputs order) ----
    const float* x          = (const float*)d_in[0];
    const float* y          = (const float*)d_in[1];
    const float* enc_ln1_g  = (const float*)d_in[2];
    const float* enc_ln1_b  = (const float*)d_in[3];
    const float* enc_ln2_g  = (const float*)d_in[4];
    const float* enc_ln2_b  = (const float*)d_in[5];
    const float* enc_wq     = (const float*)d_in[6];
    const float* enc_wk     = (const float*)d_in[7];
    const float* enc_wv     = (const float*)d_in[8];
    const float* enc_wu     = (const float*)d_in[9];
    const float* enc_bu     = (const float*)d_in[10];
    const float* enc_ffw1   = (const float*)d_in[11];
    const float* enc_ffb1   = (const float*)d_in[12];
    const float* enc_ffw2   = (const float*)d_in[13];
    const float* enc_ffb2   = (const float*)d_in[14];
    const float* enc_norm_g = (const float*)d_in[15];
    const float* enc_norm_b = (const float*)d_in[16];
    const float* learn_w    = (const float*)d_in[17];
    const float* learn_b    = (const float*)d_in[18];
    const float* dec_ln1_g  = (const float*)d_in[19];
    const float* dec_ln1_b  = (const float*)d_in[20];
    const float* dec_ln2_g  = (const float*)d_in[21];
    const float* dec_ln2_b  = (const float*)d_in[22];
    const float* dec_ln3_g  = (const float*)d_in[23];
    const float* dec_ln3_b  = (const float*)d_in[24];
    const float* dec1_wq    = (const float*)d_in[25];
    const float* dec1_wk    = (const float*)d_in[26];
    const float* dec1_wv    = (const float*)d_in[27];
    const float* dec1_wu    = (const float*)d_in[28];
    const float* dec1_bu    = (const float*)d_in[29];
    const float* dec2_wq    = (const float*)d_in[30];
    const float* dec2_wk    = (const float*)d_in[31];
    const float* dec2_wv    = (const float*)d_in[32];
    const float* dec2_wu    = (const float*)d_in[33];
    const float* dec2_bu    = (const float*)d_in[34];
    const float* dec_ffw1   = (const float*)d_in[35];
    const float* dec_ffb1   = (const float*)d_in[36];
    const float* dec_ffw2   = (const float*)d_in[37];
    const float* dec_ffb2   = (const float*)d_in[38];
    const float* dec_norm_g = (const float*)d_in[39];
    const float* dec_norm_b = (const float*)d_in[40];
    const float* map_w      = (const float*)d_in[41];
    const float* map_b      = (const float*)d_in[42];
    float* out = (float*)d_out;

    // ---- workspace carve ----
    char* ws = (char*)d_ws;
    size_t cur = 0;
    auto carve = [&](size_t bytes) -> void* {
        void* p = ws + cur;
        cur = (cur + bytes + 255) & ~(size_t)255;
        return p;
    };
    const long long nWqkv = (long long)NLAY * DE * HDE;      // 8.4M each
    const long long nWu   = (long long)NLAY * HDE * DE;
    const long long nF1   = (long long)NLAY * DE * FFD;
    const long long nF2   = (long long)NLAY * FFD * DE;
    const long long nLw   = (long long)CTXS * DE * PREDS;    // 16.8M

    __bf16* wqB  = (__bf16*)carve(nWqkv * 2);
    __bf16* wkB  = (__bf16*)carve(nWqkv * 2);
    __bf16* wvB  = (__bf16*)carve(nWqkv * 2);
    __bf16* wuB  = (__bf16*)carve(nWu * 2);
    __bf16* f1wB = (__bf16*)carve(nF1 * 2);
    __bf16* f2wB = (__bf16*)carve(nF2 * 2);
    __bf16* lwB  = (__bf16*)carve(nLw * 2);

    float*  h     = (float*)carve((long long)MROWS * DE * 4);
    __bf16* h2b   = (__bf16*)carve((long long)MROWS * DE * 2);       // LN out / enc_out flat
    __bf16* qb    = (__bf16*)carve((long long)MROWS * HDE * 2);      // also attn-out (aliased)
    __bf16* kb    = (__bf16*)carve((long long)MROWS * HDE * 2);
    __bf16* vb    = (__bf16*)carve((long long)MROWS * HDE * 2);
    float*  scores= (float*)carve((long long)BATCH * NHEAD * CTXS * CTXS * 4); // aliased f1
    __bf16* pb    = (__bf16*)carve((long long)BATCH * NHEAD * CTXS * CTXS * 2);
    __bf16* attnb = qb;                  // reuse: Q dead after scores
    __bf16* f1b   = (__bf16*)scores;     // reuse: scores dead after softmax

    float* partC  = (float*)carve(BATCH * PREDS * 4);
    float* encmap = (float*)carve(BATCH * PREDS * 4);
    float* dbuf   = (float*)carve(BATCH * PREDS * 4);
    float* d2buf  = (float*)carve(BATCH * PREDS * 4);
    float* decout = (float*)carve(BATCH * PREDS * 4);

    // ---- weight conversion fp32 -> bf16 ----
    launch_cvt(stream, enc_wq,   wqB,  nWqkv);
    launch_cvt(stream, enc_wk,   wkB,  nWqkv);
    launch_cvt(stream, enc_wv,   wvB,  nWqkv);
    launch_cvt(stream, enc_wu,   wuB,  nWu);
    launch_cvt(stream, enc_ffw1, f1wB, nF1);
    launch_cvt(stream, enc_ffw2, f2wB, nF2);
    launch_cvt(stream, learn_w,  lwB,  nLw);

    // ---- encoder ----
    pe_add_kernel<<<4096, 256, 0, stream>>>(x, h, CTXS, DE, (long long)MROWS * DE);

    const long long sQ  = (long long)CTXS * HDE;   // (b) stride in [B*S, H*D]
    const long long sSc = (long long)CTXS * CTXS;  // per-head score block

    for (int l = 0; l < NLAY; ++l) {
        const __bf16* wq_l = wqB  + (long long)l * DE * HDE;
        const __bf16* wk_l = wkB  + (long long)l * DE * HDE;
        const __bf16* wv_l = wvB  + (long long)l * DE * HDE;
        const __bf16* wu_l = wuB  + (long long)l * HDE * DE;
        const __bf16* w1_l = f1wB + (long long)l * DE * FFD;
        const __bf16* w2_l = f2wB + (long long)l * FFD * DE;

        // LN1 -> bf16
        ln_bf16_kernel<<<MROWS, 256, 0, stream>>>(h, enc_ln1_g + l * DE, enc_ln1_b + l * DE, h2b, DE);
        // Q, K, V projections (no bias)
        launch_gemm(stream, h2b, wq_l, nullptr, nullptr, nullptr, qb,
                    MROWS, HDE, DE, DE, HDE, HDE, 1, 1, 0,0, 0,0, 0,0, 0, 0, 1);
        launch_gemm(stream, h2b, wk_l, nullptr, nullptr, nullptr, kb,
                    MROWS, HDE, DE, DE, HDE, HDE, 1, 1, 0,0, 0,0, 0,0, 0, 0, 1);
        launch_gemm(stream, h2b, wv_l, nullptr, nullptr, nullptr, vb,
                    MROWS, HDE, DE, DE, HDE, HDE, 1, 1, 0,0, 0,0, 0,0, 0, 0, 1);
        // scores[b,h] = Q_bh @ K_bh^T  (no 1/sqrt(d) scaling, per reference)
        launch_gemm(stream, qb, kb, nullptr, nullptr, scores, nullptr,
                    CTXS, CTXS, DE, HDE, HDE, CTXS,
                    BATCH * NHEAD, NHEAD,
                    sQ, DE, sQ, DE, (long long)NHEAD * sSc, sSc, 0, /*transB=*/1, 1);
        // softmax rows -> bf16 probs
        softmax_bf16_kernel<<<BATCH * NHEAD * CTXS, 256, 0, stream>>>(scores, pb, CTXS);
        // attn[b,h] = P_bh @ V_bh  (written head-interleaved into [B*S, H*D])
        launch_gemm(stream, pb, vb, nullptr, nullptr, nullptr, attnb,
                    CTXS, DE, CTXS, CTXS, HDE, HDE,
                    BATCH * NHEAD, NHEAD,
                    (long long)NHEAD * sSc, sSc, sQ, DE, sQ, DE, 0, 0, 1);
        // output projection + bias + residual into h (fp32)
        launch_gemm(stream, attnb, wu_l, enc_bu + l * DE, h, h, nullptr,
                    MROWS, DE, HDE, HDE, DE, DE, 1, 1, 0,0, 0,0, 0,0, 0, 0, 1);
        // LN2 -> bf16
        ln_bf16_kernel<<<MROWS, 256, 0, stream>>>(h, enc_ln2_g + l * DE, enc_ln2_b + l * DE, h2b, DE);
        // FFN
        launch_gemm(stream, h2b, w1_l, enc_ffb1 + l * FFD, nullptr, nullptr, f1b,
                    MROWS, FFD, DE, DE, FFD, FFD, 1, 1, 0,0, 0,0, 0,0, /*relu=*/1, 0, 1);
        launch_gemm(stream, f1b, w2_l, enc_ffb2 + l * DE, h, h, nullptr,
                    MROWS, DE, FFD, FFD, DE, DE, 1, 1, 0,0, 0,0, 0,0, 0, 0, 1);
    }
    // final encoder norm -> bf16 (also the flattened [B, CTX*DE] learn input)
    ln_bf16_kernel<<<MROWS, 256, 0, stream>>>(h, enc_norm_g, enc_norm_b, h2b, DE);

    // ---- learn: [16, 262144] @ [262144, 64], split-K over 64 chunks ----
    (void)hipMemsetAsync(partC, 0, BATCH * PREDS * sizeof(float), stream);
    launch_gemm(stream, h2b, lwB, nullptr, nullptr, partC, nullptr,
                BATCH, PREDS, CTXS * DE, CTXS * DE, PREDS, PREDS,
                1, 1, 0,0, 0,0, 0,0, 0, 0, /*splitK=*/64);
    bias_relu_kernel<<<4, 256, 0, stream>>>(partC, learn_b, encmap, BATCH * PREDS, PREDS);

    // ---- decoder (d_model = 1): scalar kernels ----
    pe_add_kernel<<<4, 256, 0, stream>>>(y, dbuf, PREDS, 1, BATCH * PREDS);
    for (int l = 0; l < NLAY; ++l) {
        ln_small_kernel<<<4, 256, 0, stream>>>(dbuf, dec_ln1_g + l, dec_ln1_b + l, d2buf, BATCH * PREDS, 1);
        dec_attn_kernel<<<4, 256, 0, stream>>>(d2buf, d2buf,
            dec1_wq + l * 8, dec1_wk + l * 8, dec1_wv + l * 8, dec1_wu + l * 8,
            dec1_bu + l, dbuf, /*causal=*/1);
        ln_small_kernel<<<4, 256, 0, stream>>>(dbuf, dec_ln2_g + l, dec_ln2_b + l, d2buf, BATCH * PREDS, 1);
        dec_attn_kernel<<<4, 256, 0, stream>>>(d2buf, encmap,
            dec2_wq + l * 8, dec2_wk + l * 8, dec2_wv + l * 8, dec2_wu + l * 8,
            dec2_bu + l, dbuf, /*causal=*/0);
        ln_small_kernel<<<4, 256, 0, stream>>>(dbuf, dec_ln3_g + l, dec_ln3_b + l, d2buf, BATCH * PREDS, 1);
        dec_ffn_kernel<<<4, 256, 0, stream>>>(d2buf,
            dec_ffw1 + l * 10, dec_ffb1 + l * 10, dec_ffw2 + l * 10, dec_ffb2 + l, dbuf);
    }
    ln_small_kernel<<<4, 256, 0, stream>>>(dbuf, dec_norm_g, dec_norm_b, decout, BATCH * PREDS, 1);
    map_head_kernel<<<4, 256, 0, stream>>>(decout, map_w, map_b, out);
}
